// LSTM_46488726012494
// MI455X (gfx1250) — matrix operand order repaired and checked by reading, not compiled, not run on hardware
//
#include <hip/hip_runtime.h>
#include <hip/hip_bf16.h>
#include <math.h>

typedef __attribute__((ext_vector_type(16))) _Float16 v16h;
typedef __attribute__((ext_vector_type(8)))  _Float16 v8h;
typedef __attribute__((ext_vector_type(4)))  _Float16 h4;
typedef __attribute__((ext_vector_type(8)))  float    v8f;
typedef __attribute__((ext_vector_type(4)))  float    f4;

#ifndef __has_builtin
#define __has_builtin(x) 0
#endif
#if __has_builtin(__builtin_amdgcn_tensor_load_to_lds)
#define HAVE_TDM 1
typedef unsigned int tdm_u32x4 __attribute__((ext_vector_type(4)));
typedef int          tdm_i32x8 __attribute__((ext_vector_type(8)));
typedef int          tdm_i32x4 __attribute__((ext_vector_type(4)));
#else
#define HAVE_TDM 0
#endif

#define DEV __device__ __forceinline__

DEV float elu_f(float x)  { return x > 0.f ? x : (__expf(x) - 1.f); }
DEV float sigm_f(float x) { return 1.f / (1.f + __expf(-x)); }

// Sizes
#define TT     2048
#define HID    256
#define FEAT   1152
#define ZCOLS  1024

// ---- packed weight sizes (halves) ----
#define WXPK_HALVES (36 * 64 * 512)   // [kc=36][nt=64][lane=32][j=16]
#define WHPK_HALVES (64 * 8 * 512)    // [nt=64][kc=8][lane=32][j=16]
#define CPK_OFF0 0                    // layer0: 2 chunks -> 2048
#define CPK_OFF1 2048                 // layers 1-3: 9 chunks -> 9216 each
#define CPK_OFF2 11264
#define CPK_OFF3 20480
#define CPK_TOTAL 29696               // halves = 59,392 bytes

// ws layout (bytes)
#define WS_WXPK   0                   // 2,359,296
#define WS_WHPK   2359296             //   524,288
#define WS_CWPK   2883584             //    59,392
#define WS_FEATS  2942976             // 4,718,592
#define WS_ZX     7661568             // 8,388,608 -> end 16,050,176

// LDS layout for fused conv (halves); activation buffers have 1-px zero halo
#define LDS_W    0                    // 29,696
#define LDS_X    29696                // 86*86*4  = 29,584
#define LDS_B1   59280                // 44*44*32 = 61,952
#define LDS_B2   121232               // 23*23*32 = 16,928
#define LDS_B3   138160               // 13*13*32 =  5,408
#define LDS_HALVES 143568             // * 2 = 287,136 bytes

// ---------------------------------------------------------------------------
// K0: convert + pack weights f32 -> f16 WMMA-fragment order
// ---------------------------------------------------------------------------
DEV void pack_conv(const float* __restrict__ w, _Float16* __restrict__ dst,
                   int K, int nchunks, int i0, int stride) {
  int total = nchunks * 1024;
  for (int t = i0; t < total; t += stride) {
    int j = t & 15, lane = (t >> 4) & 31, f = (t >> 9) & 1, kc = t >> 10;
    int kr = kc * 32 + (j >> 3) * 16 + (lane >> 4) * 8 + (j & 7);
    int col = f * 16 + (lane & 15);
    dst[t] = (kr < K) ? (_Float16)w[kr * 32 + col] : (_Float16)0.f;
  }
}

__global__ void k_cvt(const float* __restrict__ cw0, const float* __restrict__ cw1,
                      const float* __restrict__ cw2, const float* __restrict__ cw3,
                      const float* __restrict__ lw,
                      _Float16* __restrict__ wxpk, _Float16* __restrict__ whpk,
                      _Float16* __restrict__ cwpk) {
  int i0 = blockIdx.x * blockDim.x + threadIdx.x;
  int stride = gridDim.x * blockDim.x;
  for (int t = i0; t < WXPK_HALVES; t += stride) {
    int j = t & 15, lane = (t >> 4) & 31, nt = (t >> 9) & 63, kc = t >> 15;
    int kr = kc * 32 + (j >> 3) * 16 + (lane >> 4) * 8 + (j & 7);
    int col = nt * 16 + (lane & 15);
    wxpk[t] = (_Float16)lw[kr * 1024 + col];
  }
  for (int t = i0; t < WHPK_HALVES; t += stride) {
    int j = t & 15, lane = (t >> 4) & 31, kc = (t >> 9) & 7, nt = t >> 12;
    int kr = FEAT + kc * 32 + (j >> 3) * 16 + (lane >> 4) * 8 + (j & 7);
    int col = nt * 16 + (lane & 15);
    whpk[t] = (_Float16)lw[kr * 1024 + col];
  }
  pack_conv(cw0, cwpk + CPK_OFF0, 36,  2, i0, stride);
  pack_conv(cw1, cwpk + CPK_OFF1, 288, 9, i0, stride);
  pack_conv(cw2, cwpk + CPK_OFF2, 288, 9, i0, stride);
  pack_conv(cw3, cwpk + CPK_OFF3, 288, 9, i0, stride);
}

// ---------------------------------------------------------------------------
// Fused conv stack with zero-halo LDS buffers: all im2col taps are in-range
// by construction -> unconditional LDS loads, zero masking.
// ---------------------------------------------------------------------------
template <int cin, int pitch>
__device__ constexpr int im2col_off(int k) {
  int ky = k / (3 * cin);
  int rem = k - ky * 3 * cin;
  int kx = rem / cin;
  int ci = rem - kx * cin;
  return (ky * pitch + kx) * cin + ci;
}

template <int inH, int outH, int cin, int pad, int KL, bool GOUT>
DEV void conv_layer_t(const _Float16* __restrict__ inL,  // LDS [inH+2][inH+2][cin]
                      _Float16* __restrict__ outP,       // LDS halo buf or global
                      const _Float16* __restrict__ wpk,  // LDS packed fragments
                      const float* __restrict__ bias) {
  constexpr int pitch  = inH + 2;
  constexpr int opitch = outH + 2;
  constexpr int M  = outH * outH;
  constexpr int Mt = (M + 15) / 16;
  constexpr int Kc = (KL + 31) / 32;
  int lane = threadIdx.x & 31, wave = threadIdx.x >> 5;
  int laneHi = lane >> 4, ln = lane & 15;
  float bia0 = bias[ln], bia1 = bias[16 + ln];

  for (int mt = wave; mt < Mt; mt += 8) {
    int m = mt * 16 + ln;
    int mc = m < M ? m : M - 1;          // clamp; extra rows never stored
    int oy = mc / outH, ox = mc - oy * outH;
    // halo-shifted base of the 3x3 receptive field for this output pixel
    const _Float16* base0 = inL + ((oy * 2 - pad + 1) * pitch + (ox * 2 - pad + 1)) * cin;
    const _Float16* baseV = base0 + laneHi * 8;   // vector path includes laneHi
    v8f c0 = {}, c1 = {};
#pragma unroll
    for (int kc = 0; kc < Kc; ++kc) {
      v16h a;
      if constexpr (cin == 32) {
        // one K-chunk == one (ky,kx) tap, ci 0..31 contiguous
        int ky = kc / 3, kx = kc - ky * 3;
        const _Float16* tap = baseV + (ky * pitch + kx) * 32;
        v8h lo = *(const v8h*)(tap);
        v8h hi = *(const v8h*)(tap + 16);
        a = __builtin_shufflevector(lo, hi, 0, 1, 2, 3, 4, 5, 6, 7,
                                    8, 9, 10, 11, 12, 13, 14, 15);
      } else {
#pragma unroll
        for (int j = 0; j < 16; ++j) {
          int k0 = kc * 32 + ((j & 8) ? 16 : 0) + (j & 7);   // laneHi = 0
          int k1 = k0 + 8;                                   // laneHi = 1
          bool v0 = k0 < KL, v1 = k1 < KL;                   // compile-time
          if (!v0 && !v1) { a[j] = (_Float16)0.f; continue; }
          int off0 = v0 ? im2col_off<cin, pitch>(k0) : 0;
          int off1 = v1 ? im2col_off<cin, pitch>(k1) : 0;
          int off  = laneHi ? off1 : off0;
          if (v0 && v1) {
            a[j] = base0[off];                               // unconditional
          } else {
            bool v = laneHi ? v1 : v0;
            _Float16 val = base0[off];
            a[j] = v ? val : (_Float16)0.f;
          }
        }
      }
      v16h b0 = *(const v16h*)(wpk + kc * 1024 + lane * 16);
      v16h b1 = *(const v16h*)(wpk + kc * 1024 + 512 + lane * 16);
      c0 = __builtin_amdgcn_wmma_f32_16x16x32_f16(false, a, false, b0, (short)0, c0, false, false);
      c1 = __builtin_amdgcn_wmma_f32_16x16x32_f16(false, a, false, b1, (short)0, c1, false, false);
    }
#pragma unroll
    for (int r = 0; r < 8; ++r) {
      int mm = mt * 16 + laneHi * 8 + r;
      if (mm < M) {
        _Float16 v0 = (_Float16)elu_f(c0[r] + bia0);
        _Float16 v1 = (_Float16)elu_f(c1[r] + bia1);
        if constexpr (GOUT) {
          outP[mm * 32 + ln] = v0;
          outP[mm * 32 + 16 + ln] = v1;
        } else {
          int my = mm / outH, mx = mm - my * outH;
          int o = ((my + 1) * opitch + (mx + 1)) * 32;
          outP[o + ln] = v0;
          outP[o + 16 + ln] = v1;
        }
      }
    }
  }
}

__global__ void __launch_bounds__(256)
k_conv(const float* __restrict__ x,
       const float* __restrict__ b0, const float* __restrict__ b1,
       const float* __restrict__ b2, const float* __restrict__ b3,
       const _Float16* __restrict__ cwpkg, _Float16* __restrict__ feats) {
  extern __shared__ _Float16 sm[];
  _Float16* wpk  = sm + LDS_W;
  _Float16* xbuf = sm + LDS_X;
  _Float16* buf1 = sm + LDS_B1;
  _Float16* buf2 = sm + LDS_B2;
  _Float16* buf3 = sm + LDS_B3;
  int img = blockIdx.x;

  // Wave 0: DMA all packed conv weights (59.4 KB) into LDS via the TDM.
#if HAVE_TDM
  if (threadIdx.x < 32) {
    unsigned long long ga = (unsigned long long)(uintptr_t)cwpkg;
    unsigned int la = (unsigned int)(uintptr_t)wpk;
    tdm_u32x4 g0;
    g0.x = 1u;
    g0.y = la;
    g0.z = (unsigned int)(ga & 0xffffffffu);
    g0.w = (unsigned int)((ga >> 32) & 0x01ffffffu) | (2u << 30);
    tdm_i32x8 g1;
    g1[0] = 0x00010000;                            // wg_mask=0, data_size=1 (2B)
    g1[1] = (int)((CPK_TOTAL & 0xffffu) << 16);    // tensor_dim0 lo16
    g1[2] = (int)(1u << 16);                       // tensor_dim1 = 1
    g1[3] = (int)((CPK_TOTAL & 0xffffu) << 16);    // tile_dim0
    g1[4] = 1;                                     // tile_dim1 = 1
    g1[5] = CPK_TOTAL;                             // tensor_dim0_stride
    g1[6] = 0;
    g1[7] = 0;
    tdm_i32x4 z4 = {0, 0, 0, 0};
#if __clang_major__ >= 23
    tdm_i32x8 z8 = {0, 0, 0, 0, 0, 0, 0, 0};
    __builtin_amdgcn_tensor_load_to_lds(g0, g1, z4, z4, z8, 0);
#else
    __builtin_amdgcn_tensor_load_to_lds(g0, g1, z4, z4, 0);
#endif
    __builtin_amdgcn_s_wait_tensorcnt(0);
  }
#else
  for (int i = threadIdx.x; i < CPK_TOTAL; i += blockDim.x) wpk[i] = cwpkg[i];
#endif

  // Bulk-zero all halo activation buffers (16B stores), overlapping the DMA.
  {
    v8h z = {};
    for (int i = threadIdx.x * 8; i < (LDS_HALVES - LDS_X); i += blockDim.x * 8)
      *(v8h*)(sm + LDS_X + i) = z;
  }
  __syncthreads();

  // Stage this image f32 -> f16 into the haloed x buffer.
  const float* xg = x + (size_t)img * (84 * 84 * 4);
  for (int p = threadIdx.x; p < 84 * 84; p += blockDim.x) {
    int r = p / 84, ccol = p - r * 84;
    f4 v = *(const f4*)(xg + p * 4);
    h4 h;
    h.x = (_Float16)v.x; h.y = (_Float16)v.y;
    h.z = (_Float16)v.z; h.w = (_Float16)v.w;
    *(h4*)(xbuf + ((r + 1) * 86 + ccol + 1) * 4) = h;
  }
  __syncthreads();

  conv_layer_t<84, 42, 4, 0, 36, false>(xbuf, buf1, wpk + CPK_OFF0, b0);
  __syncthreads();
  conv_layer_t<42, 21, 32, 0, 288, false>(buf1, buf2, wpk + CPK_OFF1, b1);
  __syncthreads();
  conv_layer_t<21, 11, 32, 1, 288, false>(buf2, buf3, wpk + CPK_OFF2, b2);
  __syncthreads();
  conv_layer_t<11, 6, 32, 1, 288, true>(buf3, feats + (size_t)img * FEAT,
                                        wpk + CPK_OFF3, b3);
}

// ---------------------------------------------------------------------------
// Time-parallel x-projection: zx = feats @ Wx + b.  1 wave per 16x16 C tile.
// ---------------------------------------------------------------------------
__global__ void __launch_bounds__(256)
k_zx(const _Float16* __restrict__ feats, const _Float16* __restrict__ wxpk,
     const float* __restrict__ lb, float* __restrict__ zx) {
  int wid = (blockIdx.x * blockDim.x + threadIdx.x) >> 5;
  int lane = threadIdx.x & 31, laneHi = lane >> 4, ln = lane & 15;
  int mt = wid >> 6, nt = wid & 63;

  const _Float16* arow = feats + (size_t)(mt * 16 + ln) * FEAT + laneHi * 8;
  const _Float16* bfrag = wxpk + (size_t)nt * 512 + lane * 16;
  v8f c = {};
#pragma unroll
  for (int kc = 0; kc < 36; ++kc) {
    __builtin_prefetch(bfrag + (size_t)(kc + 2) * (64 * 512), 0, 3);
    v8h lo = *(const v8h*)(arow + kc * 32);
    v8h hi = *(const v8h*)(arow + kc * 32 + 16);
    v16h a = __builtin_shufflevector(lo, hi, 0, 1, 2, 3, 4, 5, 6, 7,
                                     8, 9, 10, 11, 12, 13, 14, 15);
    v16h b = *(const v16h*)(bfrag + (size_t)kc * (64 * 512));
    c = __builtin_amdgcn_wmma_f32_16x16x32_f16(false, a, false, b, (short)0, c, false, false);
  }
  int n = nt * 16 + ln;
  float bias = lb[n];
#pragma unroll
  for (int r = 0; r < 8; ++r) {
    int m = mt * 16 + laneHi * 8 + r;
    zx[(size_t)m * ZCOLS + n] = c[r] + bias;
  }
}

// ---------------------------------------------------------------------------
// Sequential LSTM: persistent 32-wave workgroup; Wh fragments register-
// resident for all 2048 steps. Every A row = h; C row 0 is the result.
// ---------------------------------------------------------------------------
__global__ void __launch_bounds__(1024, 1)
k_lstm(const float* __restrict__ zx, const _Float16* __restrict__ whpk,
       const float* __restrict__ c_in, const float* __restrict__ h_in,
       float* __restrict__ hs) {
  __shared__ __align__(32) _Float16 h16[HID];
  __shared__ float cs[HID];
  __shared__ float zb[ZCOLS];
  int tid = threadIdx.x, lane = tid & 31, wave = tid >> 5;
  int laneHi = lane >> 4;

  if (tid < HID) { cs[tid] = c_in[tid]; h16[tid] = (_Float16)h_in[tid]; }

  v16h B[2][8];
#pragma unroll
  for (int p = 0; p < 2; ++p) {
    int nt = wave * 2 + p;
#pragma unroll
    for (int kc = 0; kc < 8; ++kc)
      B[p][kc] = *(const v16h*)(whpk + (size_t)(((nt * 8 + kc) * 32 + lane)) * 16);
  }
  __syncthreads();

  for (int t = 0; t < TT; ++t) {
#pragma unroll
    for (int p = 0; p < 2; ++p) {
      int nt = wave * 2 + p;
      v8f c = {};
#pragma unroll
      for (int kc = 0; kc < 8; ++kc) {
        v8h lo = *(const v8h*)(h16 + kc * 32 + laneHi * 8);
        v8h hi = *(const v8h*)(h16 + kc * 32 + 16 + laneHi * 8);
        v16h a = __builtin_shufflevector(lo, hi, 0, 1, 2, 3, 4, 5, 6, 7,
                                         8, 9, 10, 11, 12, 13, 14, 15);
        c = __builtin_amdgcn_wmma_f32_16x16x32_f16(false, a, false, B[p][kc],
                                                   (short)0, c, false, false);
      }
      if (lane < 16) zb[nt * 16 + lane] = c[0] + zx[(size_t)t * ZCOLS + nt * 16 + lane];
    }
    __syncthreads();
    if (tid < HID) {
      float zi = zb[tid], zj = zb[HID + tid], zf = zb[2 * HID + tid], zo = zb[3 * HID + tid];
      float cc = cs[tid] * sigm_f(zf + 1.f) + sigm_f(zi) * tanhf(zj);
      float hh = tanhf(cc) * sigm_f(zo);
      cs[tid] = cc;
      h16[tid] = (_Float16)hh;
      hs[(size_t)t * HID + tid] = hh;
    }
    __syncthreads();
  }
}

// ---------------------------------------------------------------------------
// logits = hs @ fc_w + fc_b
// ---------------------------------------------------------------------------
__global__ void k_fc(const float* __restrict__ hs, const float* __restrict__ fw,
                     const float* __restrict__ fb, float* __restrict__ out) {
  int idx = blockIdx.x * blockDim.x + threadIdx.x;
  if (idx >= TT * 18) return;
  int t = idx / 18, o = idx - t * 18;
  const float* h = hs + (size_t)t * HID;
  float s = fb[o];
#pragma unroll 8
  for (int u = 0; u < HID; ++u) s += h[u] * fw[u * 18 + o];
  out[idx] = s;
}

// ---------------------------------------------------------------------------
extern "C" void kernel_launch(void* const* d_in, const int* in_sizes, int n_in,
                              void* d_out, int out_size, void* d_ws, size_t ws_size,
                              hipStream_t stream) {
  (void)in_sizes; (void)n_in; (void)out_size; (void)ws_size;
  const float* x   = (const float*)d_in[0];
  const float* cw0 = (const float*)d_in[1];  const float* cb0 = (const float*)d_in[2];
  const float* cw1 = (const float*)d_in[3];  const float* cb1 = (const float*)d_in[4];
  const float* cw2 = (const float*)d_in[5];  const float* cb2 = (const float*)d_in[6];
  const float* cw3 = (const float*)d_in[7];  const float* cb3 = (const float*)d_in[8];
  const float* lw  = (const float*)d_in[9];  const float* lb  = (const float*)d_in[10];
  const float* fw  = (const float*)d_in[11]; const float* fb  = (const float*)d_in[12];
  const float* c0  = (const float*)d_in[13]; const float* h0  = (const float*)d_in[14];

  char* ws = (char*)d_ws;
  _Float16* wxpk  = (_Float16*)(ws + WS_WXPK);
  _Float16* whpk  = (_Float16*)(ws + WS_WHPK);
  _Float16* cwpk  = (_Float16*)(ws + WS_CWPK);
  _Float16* feats = (_Float16*)(ws + WS_FEATS);
  float*    zx    = (float*)   (ws + WS_ZX);

  float* out = (float*)d_out;
  float* hs  = out + TT * 18;

  k_cvt <<<512, 256, 0, stream>>>(cw0, cw1, cw2, cw3, lw, wxpk, whpk, cwpk);
  k_conv<<<TT, 256, LDS_HALVES * 2, stream>>>(x, cb0, cb1, cb2, cb3, cwpk, feats);
  k_zx  <<<1024, 256, 0, stream>>>(feats, wxpk, lb, zx);
  k_lstm<<<1, 1024, 0, stream>>>(zx, whpk, c0, h0, hs);
  k_fc  <<<(TT * 18 + 255) / 256, 256, 0, stream>>>(hs, fw, fb, out);
}
